// PatchEmbedding_2997887172850
// MI455X (gfx1250) — compile-verified, matrix operand order checked by reading, not run on hardware
//
#include <hip/hip_runtime.h>

// ---------- bf16 helpers (raw-bit storage as unsigned short) ----------
typedef __attribute__((ext_vector_type(16))) __bf16 v16bf;
typedef __attribute__((ext_vector_type(8)))  float  v8f;

union FragU {
    unsigned short s[16];
    uint4          q[2];
    v16bf          v;
};

__device__ __forceinline__ unsigned short f32_to_bf16(float f) {
    unsigned int u = __float_as_uint(f);
    unsigned int r = u + 0x7FFFu + ((u >> 16) & 1u);   // round-to-nearest-even
    return (unsigned short)(r >> 16);
}
// K index held by vector element e (0..15) for lane-half h, per CDNA5 16-bit
// A/B operand layout: elems 0-7 => K = h*8 .. h*8+7 ; elems 8-15 => K = 16+h*8 ..
__device__ __forceinline__ int k_of_elem(int e, int h) {
    return (e < 8) ? (h * 8 + e) : (16 + h * 8 + (e - 8));
}

// ---------- f32 -> bf16 packer with K padding (K-order = c*KHW + kh*KW + kw) ----------
__global__ void pack_weights_kernel(const float* __restrict__ src,
                                    unsigned short* __restrict__ dst,
                                    int M, int Kreal, int Kpad) {
    int i = blockIdx.x * blockDim.x + threadIdx.x;
    if (i >= M * Kpad) return;
    int m = i / Kpad, k = i - m * Kpad;
    dst[i] = (k < Kreal) ? f32_to_bf16(src[(size_t)m * Kreal + k]) : (unsigned short)0;
}

// ---------- f32 -> bf16 packer into NHWC GEMM-A order: k' = (kh*KW+kw)*Cin + c ----------
__global__ void pack_weights_nhwc_kernel(const float* __restrict__ src,  // [M][Cin][KH][KW]
                                         unsigned short* __restrict__ dst, // [M][KH*KW*Cin]
                                         int M, int Cin, int KHW) {
    int i = blockIdx.x * blockDim.x + threadIdx.x;
    int K = Cin * KHW;
    if (i >= M * K) return;
    int m  = i / K;
    int kp = i - m * K;
    int pix = kp / Cin;           // kh*KW + kw
    int c   = kp - pix * Cin;
    dst[i] = f32_to_bf16(src[(size_t)m * K + (size_t)c * KHW + pix]);
}

// ---------- 4-level Haar packet = 16x16 2D WHT per tile ----------
// x: [B][3][224][224] f32 ; feat NHWC: [B][14][14][960] bf16 (writes channels 0..767)
__global__ void haar_kernel(const float* __restrict__ x,
                            unsigned short* __restrict__ feat) {
    __shared__ float tile[256];
    __shared__ float tmp[256];
    int blk = blockIdx.x;                 // b*3*196 + c*196 + i*14 + j
    int j = blk % 14; blk /= 14;
    int i = blk % 14; blk /= 14;
    int c = blk % 3;  int b = blk / 3;
    int t = threadIdx.x;
    int p = t >> 4, q = t & 15;
    tile[t] = x[(((size_t)b * 3 + c) * 224 + (i * 16 + p)) * 224 + (j * 16 + q)];
    __syncthreads();
    float s = 0.f;
    #pragma unroll
    for (int qq = 0; qq < 16; ++qq) {
        float val = tile[(p << 4) + qq];
        s += (__popc(q & qq) & 1) ? -val : val;
    }
    tmp[t] = s;
    __syncthreads();
    float s2 = 0.f;
    #pragma unroll
    for (int pp = 0; pp < 16; ++pp) {
        float val = tmp[(pp << 4) + q];
        s2 += (__popc(p & pp) & 1) ? -val : val;
    }
    int u = p, v = q;
    int k = 0;
    #pragma unroll
    for (int L = 0; L < 4; ++L) {
        int bL = 2 * ((u >> L) & 1) + ((v >> L) & 1);
        k |= bL << (2 * (3 - L));
    }
    int ch = k * 3 + c;
    feat[(((size_t)b * 14 + i) * 14 + j) * 960 + ch] = f32_to_bf16(s2 * 0.0625f);
}

// ---------- helpers: pack 8 f32 accumulators -> uint4 of bf16 ----------
__device__ __forceinline__ uint4 pack8_bf16(const v8f& a, const float* bias, int mmb) {
    uint4 o;
    unsigned int w[4];
    #pragma unroll
    for (int i = 0; i < 4; ++i) {
        unsigned int lo = f32_to_bf16(a[2 * i]     + bias[mmb + 2 * i]);
        unsigned int hi = f32_to_bf16(a[2 * i + 1] + bias[mmb + 2 * i + 1]);
        w[i] = lo | (hi << 16);
    }
    o.x = w[0]; o.y = w[1]; o.z = w[2]; o.w = w[3];
    return o;
}

// ---------- conv1: gather implicit-GEMM (NCHW input, Cin=3), NHWC output ----------
// Branchless B gather: unconditional load from clamped offset + v_cndmask select.
__global__ void conv_gather_wmma_kernel(
    const unsigned short* __restrict__ A,   // [M][Kpad], k = c*KHW + kh*KW + kw
    const float* __restrict__ bias,
    const unsigned short* __restrict__ X,   // [B][Cin][Hin][Win] bf16
    unsigned short* __restrict__ Ybf,       // NHWC [B][Hout][Wout][ych_tot]
    int B, int Cin, int Hin, int Win,
    int KH, int KW, int stride, int pad,
    int Hout, int Wout, int Kreal, int Kpad,
    int ych_off, int ych_tot, int ntiles, int mblocks) {

    __shared__ unsigned int tab[160];       // padded to Kpad; zero beyond Kreal
    int KHW = KH * KW;
    for (int k = threadIdx.x; k < Kpad; k += 256) {
        unsigned int pk = 0;
        if (k < Kreal) {
            int c  = k / KHW;
            int r  = k - c * KHW;
            int kh = r / KW;
            int kw = r - kh * KW;
            unsigned int doff = (unsigned int)((c * Hin + kh) * Win + kw);
            pk = (doff << 6) | ((unsigned int)kh << 3) | (unsigned int)kw;
        }
        tab[k] = pk;
    }
    __syncthreads();

    int lane = threadIdx.x & 31;
    int half = lane >> 4;
    int col  = lane & 15;
    int tileId = blockIdx.x * 8 + (threadIdx.x >> 5);
    int total = B * mblocks * ntiles;
    if (tileId >= total) return;

    int b  = tileId / (mblocks * ntiles);
    int rm = tileId - b * (mblocks * ntiles);
    int mb = rm / ntiles;
    int nt = rm - mb * ntiles;

    int Nsp = Hout * Wout;
    int n = nt * 16 + col;
    bool valid = (n < Nsp);
    int oh = valid ? (n / Wout) : 0;
    int ow = valid ? (n - oh * Wout) : 0;
    int ihb = oh * stride - pad;
    int iwb = ow * stride - pad;
    int rowBase = ihb * Win + iwb;

    const unsigned short* Xb = X + (size_t)b * Cin * Hin * Win;
    const unsigned short* Ar[4];
    #pragma unroll
    for (int mr = 0; mr < 4; ++mr)
        Ar[mr] = A + (size_t)(mb * 64 + mr * 16 + col) * Kpad;

    v8f acc[4];
    #pragma unroll
    for (int mr = 0; mr < 4; ++mr) acc[mr] = v8f{0.f,0.f,0.f,0.f,0.f,0.f,0.f,0.f};

    for (int kb = 0; kb < Kpad; kb += 32) {
        // --- branchless im2col gather: predicate -> safe offset -> select ---
        FragU ub;
        #pragma unroll
        for (int e = 0; e < 16; ++e) {
            int k = kb + k_of_elem(e, half);
            unsigned int pk = tab[k];
            int kw = (int)(pk & 7u);
            int kh = (int)((pk >> 3) & 7u);
            bool ok = valid && (k < Kreal) &&
                      ((unsigned)(ihb + kh) < (unsigned)Hin) &&
                      ((unsigned)(iwb + kw) < (unsigned)Win);
            int off = ok ? (rowBase + (int)(pk >> 6)) : 0;   // offset 0 always in-bounds
            unsigned short bits = Xb[off];
            ub.s[e] = ok ? bits : (unsigned short)0;
        }
        FragU ua[4];
        #pragma unroll
        for (int mr = 0; mr < 4; ++mr) {
            const unsigned short* ap = Ar[mr] + kb;
            ua[mr].q[0] = *(const uint4*)(ap + half * 8);
            ua[mr].q[1] = *(const uint4*)(ap + 16 + half * 8);
        }
        acc[0] = __builtin_amdgcn_wmma_f32_16x16x32_bf16(false, ua[0].v, false, ub.v, (short)0, acc[0], false, false);
        acc[1] = __builtin_amdgcn_wmma_f32_16x16x32_bf16(false, ua[1].v, false, ub.v, (short)0, acc[1], false, false);
        acc[2] = __builtin_amdgcn_wmma_f32_16x16x32_bf16(false, ua[2].v, false, ub.v, (short)0, acc[2], false, false);
        acc[3] = __builtin_amdgcn_wmma_f32_16x16x32_bf16(false, ua[3].v, false, ub.v, (short)0, acc[3], false, false);
    }

    if (!valid) return;
    size_t pixOut = (((size_t)b * Hout + oh) * Wout + ow) * ych_tot + ych_off;
    #pragma unroll
    for (int mr = 0; mr < 4; ++mr) {
        int mmb = mb * 64 + mr * 16 + ((lane < 16) ? 0 : 8);  // 8 consecutive channels
        *(uint4*)(Ybf + pixOut + mmb) = pack8_bf16(acc[mr], bias, mmb);
    }
}

// ---------- conv2/conv3: NHWC implicit-GEMM, fully vectorized B loads ----------
// A: [M][KH*KW*Cin], k = (kh*KW+kw)*Cin + c ;  X NHWC: [B][Hin][Win][Cin] (Cin % 32 == 0)
__global__ void conv_nhwc_wmma_kernel(
    const unsigned short* __restrict__ A, const float* __restrict__ bias,
    const unsigned short* __restrict__ X,
    unsigned short* __restrict__ Ybf, float* __restrict__ Yfinal,
    const float* __restrict__ positions,
    int B, int Cin, int Hin, int Win,
    int KH, int KW, int stride, int pad,
    int Hout, int Wout,
    int ych_off, int ych_tot, int ntiles, int mblocks) {

    int lane = threadIdx.x & 31;
    int half = lane >> 4;
    int col  = lane & 15;
    int tileId = blockIdx.x * 8 + (threadIdx.x >> 5);
    int total = B * mblocks * ntiles;
    if (tileId >= total) return;

    int b  = tileId / (mblocks * ntiles);
    int rm = tileId - b * (mblocks * ntiles);
    int mb = rm / ntiles;
    int nt = rm - mb * ntiles;

    int Nsp = Hout * Wout;
    int n = nt * 16 + col;
    bool valid = (n < Nsp);
    int oh = valid ? (n / Wout) : 0;
    int ow = valid ? (n - oh * Wout) : 0;
    int ihb = oh * stride - pad;
    int iwb = ow * stride - pad;

    int K = KH * KW * Cin;
    const unsigned short* Xb = X + (size_t)b * Hin * Win * Cin;
    const unsigned short* Ar[4];
    #pragma unroll
    for (int mr = 0; mr < 4; ++mr)
        Ar[mr] = A + (size_t)(mb * 64 + mr * 16 + col) * K;

    v8f acc[4];
    #pragma unroll
    for (int mr = 0; mr < 4; ++mr) acc[mr] = v8f{0.f,0.f,0.f,0.f,0.f,0.f,0.f,0.f};

    int kb = 0;
    for (int kh = 0; kh < KH; ++kh) {
        for (int kw = 0; kw < KW; ++kw) {
            int ih = ihb + kh, iw = iwb + kw;
            bool pix = valid && (unsigned)ih < (unsigned)Hin && (unsigned)iw < (unsigned)Win;
            const unsigned short* px = Xb + ((size_t)ih * Win + iw) * Cin;
            for (int cb = 0; cb < Cin; cb += 32, kb += 32) {
                FragU ub;
                if (pix) {
                    ub.q[0] = *(const uint4*)(px + cb + half * 8);        // K = cb+h*8..
                    ub.q[1] = *(const uint4*)(px + cb + 16 + half * 8);   // K = cb+16+h*8..
                } else {
                    ub.q[0] = make_uint4(0u, 0u, 0u, 0u);
                    ub.q[1] = make_uint4(0u, 0u, 0u, 0u);
                }
                FragU ua[4];
                #pragma unroll
                for (int mr = 0; mr < 4; ++mr) {
                    const unsigned short* ap = Ar[mr] + kb;
                    ua[mr].q[0] = *(const uint4*)(ap + half * 8);
                    ua[mr].q[1] = *(const uint4*)(ap + 16 + half * 8);
                }
                if (kb + 32 < K) __builtin_prefetch(Ar[0] + kb + 32, 0, 3);  // global_prefetch_b8
                acc[0] = __builtin_amdgcn_wmma_f32_16x16x32_bf16(false, ua[0].v, false, ub.v, (short)0, acc[0], false, false);
                acc[1] = __builtin_amdgcn_wmma_f32_16x16x32_bf16(false, ua[1].v, false, ub.v, (short)0, acc[1], false, false);
                acc[2] = __builtin_amdgcn_wmma_f32_16x16x32_bf16(false, ua[2].v, false, ub.v, (short)0, acc[2], false, false);
                acc[3] = __builtin_amdgcn_wmma_f32_16x16x32_bf16(false, ua[3].v, false, ub.v, (short)0, acc[3], false, false);
            }
        }
    }

    if (!valid) return;
    if (Yfinal) {
        // fused bias + positions + NCHW->NLC transpose; 8 consecutive channels per frag
        float* orow = Yfinal + ((size_t)b * 50 + 1 + n) * 768;
        const float* prow = positions + (size_t)(1 + n) * 768;
        #pragma unroll
        for (int mr = 0; mr < 4; ++mr) {
            int mmb = mb * 64 + mr * 16 + ((lane < 16) ? 0 : 8);
            float4 p0 = *(const float4*)(prow + mmb);
            float4 p1 = *(const float4*)(prow + mmb + 4);
            float4 b0 = *(const float4*)(bias + mmb);
            float4 b1 = *(const float4*)(bias + mmb + 4);
            float4 o0, o1;
            o0.x = acc[mr][0] + b0.x + p0.x;  o0.y = acc[mr][1] + b0.y + p0.y;
            o0.z = acc[mr][2] + b0.z + p0.z;  o0.w = acc[mr][3] + b0.w + p0.w;
            o1.x = acc[mr][4] + b1.x + p1.x;  o1.y = acc[mr][5] + b1.y + p1.y;
            o1.z = acc[mr][6] + b1.z + p1.z;  o1.w = acc[mr][7] + b1.w + p1.w;
            *(float4*)(orow + mmb)     = o0;
            *(float4*)(orow + mmb + 4) = o1;
        }
    } else {
        size_t pixOut = (((size_t)b * Hout + oh) * Wout + ow) * ych_tot + ych_off;
        #pragma unroll
        for (int mr = 0; mr < 4; ++mr) {
            int mmb = mb * 64 + mr * 16 + ((lane < 16) ? 0 : 8);
            *(uint4*)(Ybf + pixOut + mmb) = pack8_bf16(acc[mr], bias, mmb);
        }
    }
}

// ---------- cls row: out[b][0][oc] = cls + positions[0][oc] ----------
__global__ void cls_kernel(const float* __restrict__ cls_token,
                           const float* __restrict__ positions,
                           float* __restrict__ out, int B) {
    int i = blockIdx.x * blockDim.x + threadIdx.x;
    if (i >= B * 768) return;
    int oc = i % 768, b = i / 768;
    out[(size_t)b * 50 * 768 + oc] = cls_token[oc] + positions[oc];
}

extern "C" void kernel_launch(void* const* d_in, const int* in_sizes, int n_in,
                              void* d_out, int out_size, void* d_ws, size_t ws_size,
                              hipStream_t stream) {
    (void)in_sizes; (void)n_in; (void)out_size; (void)ws_size;
    const float* x   = (const float*)d_in[0];   // [64,3,224,224]
    const float* w1  = (const float*)d_in[1];   // [192,3,7,7]
    const float* b1  = (const float*)d_in[2];
    const float* w2  = (const float*)d_in[3];   // [192,192,7,7]
    const float* b2  = (const float*)d_in[4];
    const float* w3  = (const float*)d_in[5];   // [768,960,3,3]
    const float* b3  = (const float*)d_in[6];
    const float* cls = (const float*)d_in[7];
    const float* pos = (const float*)d_in[8];
    float* out = (float*)d_out;

    char* ws = (char*)d_ws;
    unsigned short* xb   = (unsigned short*)(ws + 0);           // 64*3*224*224 bf16 (NCHW)
    unsigned short* A1   = (unsigned short*)(ws + 19267584);    // 192*160       (c-major K)
    unsigned short* A2   = (unsigned short*)(ws + 19329024);    // 192*9408      (pixel-major K)
    unsigned short* A3   = (unsigned short*)(ws + 22941696);    // 768*8640      (pixel-major K)
    unsigned short* sp1  = (unsigned short*)(ws + 36212736);    // NHWC 64*56*56*192
    unsigned short* feat = (unsigned short*)(ws + 113283072);   // NHWC 64*14*14*960

    // prep: cast x + pack weights
    pack_weights_kernel<<<(9633792 + 255) / 256, 256, 0, stream>>>(x,  xb, 1, 9633792, 9633792);
    pack_weights_kernel<<<(192 * 160 + 255) / 256, 256, 0, stream>>>(w1, A1, 192, 147, 160);
    pack_weights_nhwc_kernel<<<(192 * 9408 + 255) / 256, 256, 0, stream>>>(w2, A2, 192, 192, 49);
    pack_weights_nhwc_kernel<<<(768 * 8640 + 255) / 256, 256, 0, stream>>>(w3, A3, 768, 960, 9);

    // wavelet branch -> feat NHWC channels [0,768)
    haar_kernel<<<64 * 3 * 14 * 14, 256, 0, stream>>>(x, feat);

    // conv1: 3->192, 7x7 s4 p3, 224->56, gather path -> sp1 NHWC. waves = 64*3*196
    conv_gather_wmma_kernel<<<4704, 256, 0, stream>>>(
        A1, b1, xb, sp1,
        64, 3, 224, 224, 7, 7, 4, 3, 56, 56, 147, 160, 0, 192, 196, 3);

    // conv2: 192->192, 7x7 s4 p3, 56->14 -> feat NHWC channels [768,960). waves = 64*3*13
    conv_nhwc_wmma_kernel<<<312, 256, 0, stream>>>(
        A2, b2, sp1, feat, nullptr, nullptr,
        64, 192, 56, 56, 7, 7, 4, 3, 14, 14, 768, 960, 13, 3);

    // conv3: 960->768, 3x3 s2 p1, 14->7, fused bias+positions+transpose. waves = 64*12*4
    conv_nhwc_wmma_kernel<<<384, 256, 0, stream>>>(
        A3, b3, feat, nullptr, out, pos,
        64, 960, 14, 14, 3, 3, 2, 1, 7, 7, 0, 0, 4, 12);

    // cls row
    cls_kernel<<<192, 256, 0, stream>>>(cls, pos, out, 64);
}
